// HierarchicalAttention_3058016715076
// MI455X (gfx1250) — compile-verified, matrix operand order checked by reading, not run on hardware
//
#include <hip/hip_runtime.h>
#include <hip/hip_bf16.h>

// Problem constants
#define BS 8
#define SS 32
#define LL 128
#define DD 768
#define PP 496
#define MROWS (BS * SS * LL)   // 32768 GEMM rows

// ---------- WMMA types (CDNA5 gfx1250, wave32) ----------
typedef __attribute__((ext_vector_type(16))) __bf16 v16bf;
typedef __attribute__((ext_vector_type(8)))  float  v8f;

union Frag {          // 16 bf16 = 32B = two 16B LDS/global loads
    uint4 u[2];
    v16bf v;
};

// ---------- fp32 -> bf16 split helpers (RNE) ----------
static __device__ __forceinline__ unsigned short f32_to_bf16(float f) {
    unsigned int u = __float_as_uint(f);
    u += 0x7fffu + ((u >> 16) & 1u);
    return (unsigned short)(u >> 16);
}
static __device__ __forceinline__ float bf16_to_f32(unsigned short h) {
    return __uint_as_float(((unsigned int)h) << 16);
}

// ============================================================
// Kernel 0: W1 (fp32, [k][n]) -> transposed bf16 hi/lo  Wt[n][k]
// B-fragment-friendly: per lane, contiguous runs of 8 bf16 in k.
// ============================================================
__global__ void __launch_bounds__(256)
wprep_kernel(const float* __restrict__ W1,
             unsigned short* __restrict__ Whi,
             unsigned short* __restrict__ Wlo) {
    int idx = blockIdx.x * 256 + threadIdx.x;     // idx = n*768 + k
    if (idx >= DD * DD) return;
    int n = idx / DD, k = idx - n * DD;
    float f = W1[k * DD + n];
    unsigned short hi = f32_to_bf16(f);
    Whi[idx] = hi;
    Wlo[idx] = f32_to_bf16(f - bf16_to_f32(hi));
}

// ============================================================
// Kernel 1: fused  scores[m] = sum_h tanh( (X W1)[m,h] + b1[h] ) * w2[h]
// bf16x3 split GEMM on v_wmma_f32_16x16x32_bf16. 64-row tile in LDS.
// ============================================================
__global__ void __launch_bounds__(256)
scores_kernel(const float* __restrict__ X,
              const unsigned short* __restrict__ Whi,
              const unsigned short* __restrict__ Wlo,
              const float* __restrict__ b1,
              const float* __restrict__ w2,
              float* __restrict__ scores) {
    extern __shared__ unsigned char smem_raw[];
    unsigned short* Ahi = (unsigned short*)smem_raw;        // 64*768 bf16 (96KB)
    unsigned short* Alo = Ahi + 64 * DD;                    // 64*768 bf16 (96KB)
    float* s_score      = (float*)(Alo + 64 * DD);          // 64 floats

    const int tid = threadIdx.x;
    const int wg  = blockIdx.x;
    const int rowBase = wg * 64;

    if (tid < 64) s_score[tid] = 0.0f;

    // Stage 64x768 fp32 tile -> bf16 hi/lo in LDS (coalesced in k)
    const float* Xt = X + (size_t)rowBase * DD;
    for (int i = tid; i < 64 * DD; i += 256) {
        float f = Xt[i];
        unsigned short hi = f32_to_bf16(f);
        Ahi[i] = hi;
        Alo[i] = f32_to_bf16(f - bf16_to_f32(hi));
    }
    __syncthreads();

    const int wave = tid >> 5;
    const int lane = tid & 31;
    const int lmod = lane & 15;          // column (B/C) or row (A) within 16
    const int kb   = (lane < 16) ? 0 : 8; // K-half per ISA 16-bit layouts

    float sacc[4][8];                    // per-lane score partials, 4 row slices
    #pragma unroll
    for (int r = 0; r < 4; ++r)
        #pragma unroll
        for (int v = 0; v < 8; ++v) sacc[r][v] = 0.0f;

    // Wave owns 6 N-tiles (16 cols each) and all 4 row slices:
    // each B fragment loaded once per workgroup -> max W1 (L2) reuse.
    for (int ci = 0; ci < 6; ++ci) {
        const int c    = wave + ci * 8;          // N-tile 0..47
        const int ncol = c * 16 + lmod;
        const unsigned short* bh_base = Whi + (size_t)ncol * DD;
        const unsigned short* bl_base = Wlo + (size_t)ncol * DD;

        v8f acc[4] = {};                         // fp32 accumulators (4 x 16x16)

        for (int k0 = 0; k0 < DD; k0 += 32) {
            Frag bhi, blo;
            bhi.u[0] = *(const uint4*)(bh_base + k0 + kb);
            bhi.u[1] = *(const uint4*)(bh_base + k0 + kb + 16);
            blo.u[0] = *(const uint4*)(bl_base + k0 + kb);
            blo.u[1] = *(const uint4*)(bl_base + k0 + kb + 16);
            #pragma unroll
            for (int r = 0; r < 4; ++r) {
                const unsigned short* ah = Ahi + (r * 16 + lmod) * DD + k0 + kb;
                const unsigned short* al = Alo + (r * 16 + lmod) * DD + k0 + kb;
                Frag ahi_f, alo_f;
                ahi_f.u[0] = *(const uint4*)(ah);
                ahi_f.u[1] = *(const uint4*)(ah + 16);
                alo_f.u[0] = *(const uint4*)(al);
                alo_f.u[1] = *(const uint4*)(al + 16);
                // bf16x3: Ah*Bh + Ah*Bl + Al*Bh  (~fp32 accuracy)
                acc[r] = __builtin_amdgcn_wmma_f32_16x16x32_bf16(
                            false, ahi_f.v, false, bhi.v, (short)0, acc[r], false, false);
                acc[r] = __builtin_amdgcn_wmma_f32_16x16x32_bf16(
                            false, ahi_f.v, false, blo.v, (short)0, acc[r], false, false);
                acc[r] = __builtin_amdgcn_wmma_f32_16x16x32_bf16(
                            false, alo_f.v, false, bhi.v, (short)0, acc[r], false, false);
            }
        }
        // Fused epilogue: +b1, tanh, *w2. C layout: lane -> N=lmod,
        // element v -> M = v + 8*(lane/16); bias/w2 scalar per lane.
        const float b1c = b1[ncol];
        const float w2c = w2[ncol];
        #pragma unroll
        for (int r = 0; r < 4; ++r)
            #pragma unroll
            for (int v = 0; v < 8; ++v)
                sacc[r][v] += tanhf(acc[r][v] + b1c) * w2c;
    }

    // Reduce over the 16 N-lanes (xor <16 stays within each 16-lane half)
    #pragma unroll
    for (int r = 0; r < 4; ++r)
        #pragma unroll
        for (int v = 0; v < 8; ++v)
            for (int off = 1; off < 16; off <<= 1)
                sacc[r][v] += __shfl_xor(sacc[r][v], off, 32);

    if (lmod == 0) {
        const int half = (lane >> 4) * 8;      // lane0 -> rows v, lane16 -> rows 8+v
        #pragma unroll
        for (int r = 0; r < 4; ++r)
            #pragma unroll
            for (int v = 0; v < 8; ++v)
                atomicAdd(&s_score[r * 16 + half + v], sacc[r][v]);
    }
    __syncthreads();
    if (tid < 64) scores[rowBase + tid] = s_score[tid];
}

// ============================================================
// Kernel 2: per-(b,s) masked softmax over L=128 + weighted pool of X
// ============================================================
__global__ void __launch_bounds__(256)
softmax_pool_kernel(const float* __restrict__ scores,
                    const float* __restrict__ mask,
                    const float* __restrict__ b2,
                    const float* __restrict__ X,
                    float* __restrict__ out) {
    __shared__ float s_val[LL];
    __shared__ float s_red[LL];
    const int bs  = blockIdx.x;        // 0..255 (b*S+s)
    const int tid = threadIdx.x;

    if (tid < LL) {
        float sc = scores[bs * LL + tid] + b2[0]
                 + (1.0f - mask[bs * LL + tid]) * -10000.0f;
        s_val[tid] = sc;
        s_red[tid] = sc;
    }
    __syncthreads();
    for (int st = 64; st > 0; st >>= 1) {
        if (tid < st) s_red[tid] = fmaxf(s_red[tid], s_red[tid + st]);
        __syncthreads();
    }
    const float mx = s_red[0];
    __syncthreads();
    if (tid < LL) {
        float e = __expf(s_val[tid] - mx);
        s_val[tid] = e;
        s_red[tid] = e;
    }
    __syncthreads();
    for (int st = 64; st > 0; st >>= 1) {
        if (tid < st) s_red[tid] += s_red[tid + st];
        __syncthreads();
    }
    const float inv = 1.0f / s_red[0];
    __syncthreads();
    if (tid < LL) s_val[tid] *= inv;
    __syncthreads();

    const float* Xb = X + (size_t)bs * LL * DD;
    float a0 = 0.f, a1 = 0.f, a2 = 0.f;
    for (int l = 0; l < LL; ++l) {
        const float p = s_val[l];
        const float* row = Xb + l * DD;
        a0 += p * row[tid];
        a1 += p * row[tid + 256];
        a2 += p * row[tid + 512];
    }
    out[bs * DD + tid]       = a0;
    out[bs * DD + tid + 256] = a1;
    out[bs * DD + tid + 512] = a2;
}

// ============================================================
// Kernel 3: zero the (BS,S,S,D) output block (float4 stores)
// ============================================================
__global__ void __launch_bounds__(256)
zero4_kernel(float4* __restrict__ p, int n4) {
    int i = blockIdx.x * 256 + threadIdx.x;
    if (i < n4) p[i] = make_float4(0.f, 0.f, 0.f, 0.f);
}

// ============================================================
// Kernel 4: scatter x_cls rows (per-batch indices are unique -> plain store)
// ============================================================
__global__ void __launch_bounds__(256)
scatter_kernel(const float* __restrict__ x_cls,
               const int* __restrict__ pairs_list,
               const int* __restrict__ pairs_num,
               float* __restrict__ out2) {
    const int blk = blockIdx.x;
    const int b = blk / PP;
    const int p = blk - b * PP;
    if (p >= pairs_num[b]) return;
    const int i = pairs_list[(b * PP + p) * 2 + 0];
    const int j = pairs_list[(b * PP + p) * 2 + 1];
    const float* src = x_cls + (size_t)(b * PP + p) * DD;
    float* dst = out2 + (size_t)((b * SS + i) * SS + j) * DD;
    const int tid = threadIdx.x;
    dst[tid]       = src[tid];
    dst[tid + 256] = src[tid + 256];
    dst[tid + 512] = src[tid + 512];
}

// ============================================================
extern "C" void kernel_launch(void* const* d_in, const int* in_sizes, int n_in,
                              void* d_out, int out_size, void* d_ws, size_t ws_size,
                              hipStream_t stream) {
    const float* x_single   = (const float*)d_in[0];
    const float* x_cls      = (const float*)d_in[1];
    const float* mask       = (const float*)d_in[2];
    const float* W1         = (const float*)d_in[3];
    const float* b1         = (const float*)d_in[4];
    const float* w2         = (const float*)d_in[5];
    const float* b2         = (const float*)d_in[6];
    const int*   pairs_list = (const int*)d_in[7];
    const int*   pairs_num  = (const int*)d_in[8];

    // Workspace: Whi (1.18MB) | Wlo (1.18MB) | scores (128KB)  ~= 2.5MB
    unsigned short* Whi = (unsigned short*)d_ws;
    unsigned short* Wlo = Whi + (size_t)DD * DD;
    float* scores       = (float*)(Wlo + (size_t)DD * DD);

    float* final_sent = (float*)d_out;                  // (BS,S,D)
    float* cls_mat    = final_sent + BS * SS * DD;      // (BS,S,S,D)

    // 0) W1 -> transposed bf16 hi/lo
    wprep_kernel<<<(DD * DD + 255) / 256, 256, 0, stream>>>(W1, Whi, Wlo);

    // 1) fused GEMM+tanh+dot(w2): 512 WGs x 8 waves, 192KB dynamic LDS
    const size_t ldsA = (size_t)2 * 64 * DD * sizeof(unsigned short)
                      + 64 * sizeof(float);
    (void)hipFuncSetAttribute((const void*)scores_kernel,
                              hipFuncAttributeMaxDynamicSharedMemorySize,
                              (int)ldsA);
    scores_kernel<<<MROWS / 64, 256, ldsA, stream>>>(x_single, Whi, Wlo, b1, w2, scores);

    // 2) softmax + pooling -> final_sent
    softmax_pool_kernel<<<BS * SS, 256, 0, stream>>>(scores, mask, b2, x_single, final_sent);

    // 3) zero + 4) scatter -> cls_output_matrix_nn
    const int n4 = BS * SS * SS * DD / 4;
    zero4_kernel<<<(n4 + 255) / 256, 256, 0, stream>>>((float4*)cls_mat, n4);
    scatter_kernel<<<BS * PP, 256, 0, stream>>>(x_cls, pairs_list, pairs_num, cls_mat);
}
